// TiedAxialSelfAttention2d_88106959110285
// MI455X (gfx1250) — matrix operand order repaired and checked by reading, not compiled, not run on hardware
//
#include <hip/hip_runtime.h>
#include <hip/hip_bf16.h>
#include <stdint.h>

// ---------------------------------------------------------------------------
// TiedAxialSelfAttention2d for MI455X (gfx1250), wave32.
// Matrix math: v_wmma_f32_16x16x32_f16 (f32 accumulate).
// Data movement: GLOBAL_LOAD_ASYNC_TO_LDS_B128 double-buffered staging for the
// two big QKV GEMMs (ASYNCcnt), TENSOR_LOAD_TO_LDS (TDM, TENSORcnt) for the
// attention-output V tiles. Workspace requirement: ~235 MB.
// ---------------------------------------------------------------------------

typedef _Float16 half_t;
typedef __attribute__((ext_vector_type(8)))  _Float16 v8h;
typedef __attribute__((ext_vector_type(16))) _Float16 v16h;
typedef __attribute__((ext_vector_type(8)))  float    v8f;
typedef __attribute__((ext_vector_type(4)))  uint32_t u32x4;
typedef __attribute__((ext_vector_type(8)))  uint32_t u32x8;

#define H_   12
#define DH_  64
#define D_   768
#define TD_  2304          // 3*D
#define E_   64
#define L_   256
#define N_   16384         // E*L rows
#define NEGV (-10000.0f)
#define EPS_ 1e-5f

// ---- WMMA fragment helpers (CDNA5 ISA 7.12.2 layouts) ----------------------

__device__ __forceinline__ v16h cat8(v8h lo, v8h hi) {
  v16h r;
#pragma unroll
  for (int i = 0; i < 8; ++i) { r[i] = lo[i]; r[i + 8] = hi[i]; }
  return r;
}

// A: 16(M) x 32(K), source row-major (K contiguous), row stride ld elements.
// lanes 0-15 hold row m=lane, K={0..7,16..23}; lanes 16-31 K={8..15,24..31}.
__device__ __forceinline__ v16h load_a_frag(const half_t* p0, int ld, int lane) {
  const int m  = lane & 15;
  const int k0 = (lane >> 4) * 8;
  const half_t* p = p0 + (size_t)m * ld + k0;
  return cat8(*(const v8h*)p, *(const v8h*)(p + 16));
}

// B: 32(K) x 16(N), source stored as [n rows][k cols] (K contiguous), row
// stride ld. lanes 0-15 hold col n=lane, K={0..7,16..23}; lanes 16-31 K={8..15,24..31}.
__device__ __forceinline__ v16h load_b_frag(const half_t* p0, int ld, int lane) {
  const int n  = lane & 15;
  const int k0 = (lane >> 4) * 8;
  const half_t* p = p0 + (size_t)n * ld + k0;
  return cat8(*(const v8h*)p, *(const v8h*)(p + 16));
}

__device__ __forceinline__ v8f wmma32(v16h a, v16h b, v8f c) {
  return __builtin_amdgcn_wmma_f32_16x16x32_f16(
      /*neg_a=*/false, a, /*neg_b=*/false, b,
      /*c_mod=*/(short)0, c, /*reuse_a=*/false, /*reuse_b=*/false);
}

// ---- CDNA5 async / TDM data movers ----------------------------------------

// Per-lane async copy of 16 bytes global -> LDS (tracked by ASYNCcnt).
__device__ __forceinline__ void async_ld_b128(uint32_t lds_byte_addr,
                                              const half_t* g) {
  asm volatile("global_load_async_to_lds_b128 %0, %1, off"
               :: "v"(lds_byte_addr), "v"((uint64_t)(uintptr_t)g)
               : "memory");
}

__device__ __forceinline__ void wait_async_le4() {
  asm volatile("s_wait_asynccnt 0x4" ::: "memory");
}
__device__ __forceinline__ void wait_async_0() {
  asm volatile("s_wait_asynccnt 0x0" ::: "memory");
}

// TDM: DMA a 2D tile (tile_dim0=32 contiguous halves, tile_dim1=768 rows,
// row stride 16384 halves) from global into LDS (tracked by TENSORcnt).
__device__ __forceinline__ void tdm_load_vt(uint32_t lds_byte_addr,
                                            const half_t* g) {
  const uint64_t ga = (uint64_t)(uintptr_t)g;
  u32x4 g0;
  g0[0] = 1u;                                   // count=1, user descriptor
  g0[1] = lds_byte_addr;                        // lds_addr [63:32]
  g0[2] = (uint32_t)ga;                         // global_addr low
  g0[3] = (uint32_t)((ga >> 32) & 0x01FFFFFFu)  // global_addr[56:32]
          | 0x80000000u;                        // type=2 ("image") [127:126]
  u32x8 g1;
  g1[0] = 1u << 16;          // wg_mask=0, data_size=1 (2 bytes)
  g1[1] = 0x4000u << 16;     // tensor_dim0 = 16384 (low 16 bits at [63:48])
  g1[2] = 768u << 16;        // tensor_dim0 hi=0 | tensor_dim1 = 768
  g1[3] = 32u << 16;         // tensor_dim1 hi=0 | tile_dim0 = 32
  g1[4] = 768u;              // tile_dim1 = 768, tile_dim2 = 0
  g1[5] = 16384u;            // tensor_dim0_stride low32
  g1[6] = 0u;                // stride hi / dim1_stride (unused, 2D tile)
  g1[7] = 0u;
  asm volatile("tensor_load_to_lds %0, %1" :: "s"(g0), "s"(g1) : "memory");
}

// ---- utility kernels -------------------------------------------------------

__global__ __launch_bounds__(256) void k_f32_to_f16(const float* __restrict__ src,
                                                    half_t* __restrict__ dst, int n) {
  int i = blockIdx.x * 256 + threadIdx.x;
  if (i < n) dst[i] = (half_t)src[i];
}

__global__ __launch_bounds__(256) void k_masksum(const uint8_t* __restrict__ pm,
                                                 float* __restrict__ msum) {
  int l = threadIdx.x;  // one block of 256
  float s = 0.f;
  for (int e = 0; e < E_; ++e) s += pm[e * L_ + l] ? NEGV : 0.f;
  msum[l] = s;
}

// ---- QKV projection GEMM: C[N x 2304] = A[N x 768] * W[2304 x 768]^T + bias.
// Block tile 128x128; A/W K-slabs (128x32 f16, 8KB each) staged into LDS via
// async-to-LDS copies, double-buffered on ASYNCcnt. 8 waves each own 32x64.
// If storeVT, the V columns (>=1536) are written transposed: vT[d][n].
__global__ __launch_bounds__(256) void k_qkv_gemm(
    const half_t* __restrict__ A, const half_t* __restrict__ W,
    const float* __restrict__ bias, half_t* __restrict__ C,
    half_t* __restrict__ vT, int storeVT) {
  __shared__ half_t sA[2][128 * 32];
  __shared__ half_t sW[2][128 * 32];
  const int lane = threadIdx.x & 31;
  const int wv   = threadIdx.x >> 5;
  const int row0 = blockIdx.y * 128;
  const int col0 = blockIdx.x * 128;
  const int wr   = (wv >> 1) * 32;   // wave row offset within block tile
  const int wc   = (wv & 1) * 64;    // wave col offset within block tile

  const uint32_t ldsA = (uint32_t)(uintptr_t)&sA[0][0];
  const uint32_t ldsW = (uint32_t)(uintptr_t)&sW[0][0];
  const int r0 = threadIdx.x >> 2;          // staging rows (two per thread)
  const int r1 = r0 + 64;
  const int sg = (threadIdx.x & 3) * 8;     // 8-half segment within row

  // stage K-slab k0 into buffer buf: 4 async b128 copies per thread
  auto stage = [&](int buf, int k0) {
    const uint32_t la = ldsA + (uint32_t)buf * (128 * 32 * 2);
    const uint32_t lw = ldsW + (uint32_t)buf * (128 * 32 * 2);
    async_ld_b128(la + (uint32_t)(r0 * 32 + sg) * 2,
                  A + (size_t)(row0 + r0) * D_ + k0 + sg);
    async_ld_b128(la + (uint32_t)(r1 * 32 + sg) * 2,
                  A + (size_t)(row0 + r1) * D_ + k0 + sg);
    async_ld_b128(lw + (uint32_t)(r0 * 32 + sg) * 2,
                  W + (size_t)(col0 + r0) * D_ + k0 + sg);
    async_ld_b128(lw + (uint32_t)(r1 * 32 + sg) * 2,
                  W + (size_t)(col0 + r1) * D_ + k0 + sg);
  };

  v8f acc[2][4] = {};
  stage(0, 0);
  for (int k0 = 0, it = 0; k0 < D_; k0 += 32, ++it) {
    const int buf = it & 1;
    if (k0 + 32 < D_) { stage(buf ^ 1, k0 + 32); wait_async_le4(); }
    else              { wait_async_0(); }
    __syncthreads();   // all waves' async writes to buf are now visible

    const half_t* pa = &sA[buf][0];
    const half_t* pw = &sW[buf][0];
    v16h a0 = load_a_frag(pa + (wr + 0)  * 32, 32, lane);
    v16h a1 = load_a_frag(pa + (wr + 16) * 32, 32, lane);
#pragma unroll
    for (int j = 0; j < 4; ++j) {
      v16h b = load_b_frag(pw + (wc + 16 * j) * 32, 32, lane);
      acc[0][j] = wmma32(a0, b, acc[0][j]);
      acc[1][j] = wmma32(a1, b, acc[1][j]);
    }
    __syncthreads();   // done reading buf before it is re-staged
  }

  const int n  = lane & 15;
  const int m0 = (lane >> 4) * 8;
#pragma unroll
  for (int i = 0; i < 2; ++i)
#pragma unroll
    for (int j = 0; j < 4; ++j) {
      const int c  = col0 + wc + 16 * j + n;
      const float bv = bias[c];
#pragma unroll
      for (int r = 0; r < 8; ++r) {
        const int g = row0 + wr + 16 * i + m0 + r;
        const float v = acc[i][j][r] + bv;
        if (storeVT && c >= 2 * D_) vT[(size_t)(c - 2 * D_) * N_ + g] = (half_t)v;
        else                        C[(size_t)g * TD_ + c] = (half_t)v;
      }
    }
}

// ---- Tied row attention logits + softmax.
// Block = (i-tile of 16, head h). logits[i,j] = sum_{s,c} q[s,i,h,c]k[s,j,h,c].
// 8 waves x 2 j-tiles cover j=0..255; K-loop = 64 seqs x 2 chunks = 128 steps.
__global__ __launch_bounds__(256) void k_row_attn_probs(
    const half_t* __restrict__ qkv, const float* __restrict__ msum,
    float* __restrict__ maps_f32, half_t* __restrict__ maps_f16) {
  __shared__ float sm[16][L_];
  __shared__ float red[16][16];
  const int lane = threadIdx.x & 31;
  const int wv   = threadIdx.x >> 5;
  const int i0   = blockIdx.x * 16;
  const int h    = blockIdx.y;

  v8f acc[2] = {};
  for (int s = 0; s < E_; ++s) {
    const half_t* qb = qkv + (size_t)(s * L_ + i0) * TD_ + h * DH_;
    if (s + 1 < E_)  // hint next sequence's Q-block toward the caches
      __builtin_prefetch(qkv + (size_t)((s + 1) * L_ + i0) * TD_ + h * DH_, 0, 1);
#pragma unroll
    for (int kc = 0; kc < DH_; kc += 32) {
      v16h a = load_a_frag(qb + kc, TD_, lane);
#pragma unroll
      for (int u = 0; u < 2; ++u) {
        const int j0 = (wv + 8 * u) * 16;
        const half_t* kb = qkv + (size_t)(s * L_ + j0) * TD_ + D_ + h * DH_ + kc;
        acc[u] = wmma32(a, load_b_frag(kb, TD_, lane), acc[u]);
      }
    }
  }
  const float scale = 1.0f / 64.0f;  // (DH*E)^-0.5
  const int n = lane & 15, m0 = (lane >> 4) * 8;
#pragma unroll
  for (int u = 0; u < 2; ++u) {
    const int j0 = (wv + 8 * u) * 16;
    const float ms = msum[j0 + n];
#pragma unroll
    for (int r = 0; r < 8; ++r) sm[m0 + r][j0 + n] = acc[u][r] * scale + ms;
  }
  __syncthreads();

  // softmax over j=256 per row; 16 threads per row, 16 cols each
  const int row = threadIdx.x >> 4, seg = threadIdx.x & 15;
  float lm = -3.0e38f;
  for (int c = 0; c < 16; ++c) lm = fmaxf(lm, sm[row][seg * 16 + c]);
  red[row][seg] = lm; __syncthreads();
  float mx = -3.0e38f;
  for (int k = 0; k < 16; ++k) mx = fmaxf(mx, red[row][k]);
  __syncthreads();
  float ls = 0.f;
  for (int c = 0; c < 16; ++c) {
    float e = __expf(sm[row][seg * 16 + c] - mx);
    sm[row][seg * 16 + c] = e; ls += e;
  }
  red[row][seg] = ls; __syncthreads();
  float tot = 0.f;
  for (int k = 0; k < 16; ++k) tot += red[row][k];
  const float inv = 1.0f / tot;
  const size_t base = (size_t)h * L_ * L_ + (size_t)(i0 + row) * L_;
  for (int c = 0; c < 16; ++c) {
    const float p = sm[row][seg * 16 + c] * inv;
    maps_f32[base + seg * 16 + c] = p;       // output #2 (row_maps, fp32)
    maps_f16[base + seg * 16 + c] = (half_t)p;
  }
}

// ---- row_out = P @ V (via vT), + residual + LayerNorm1, fused.
// Block = (i-tile of 16, seq s). K-outer over j; per K-step one TDM descriptor
// DMAs the 768x32 vT slab into LDS (48KB); 48 (h,ct) tiles over 8 waves.
// The V slab and the LayerNorm staging buffer are time-disjoint -> union.
union RowOutSmem {
  half_t sV[768 * 32];   // vT[d][j0..j0+32) slab
  float  yt[16][D_];     // residual + LN staging
};

__global__ __launch_bounds__(256) void k_row_out_ln1(
    const half_t* __restrict__ P, const half_t* __restrict__ vT,
    const float* __restrict__ x, const float* __restrict__ g1,
    const float* __restrict__ b1, float* __restrict__ y,
    half_t* __restrict__ yh) {
  __shared__ RowOutSmem us;
  __shared__ float red[16][16];
  const int lane = threadIdx.x & 31;
  const int wv   = threadIdx.x >> 5;
  const int i0   = blockIdx.x * 16;
  const int s    = blockIdx.y;
  const int n = lane & 15, m0 = (lane >> 4) * 8;
  const uint32_t ldsV = (uint32_t)(uintptr_t)&us.sV[0];

  v8f acc[6] = {};
  for (int j0 = 0; j0 < L_; j0 += 32) {
    if (threadIdx.x < 32) {                       // wave 0 drives the TDM
      tdm_load_vt(ldsV, vT + s * L_ + j0);
      __builtin_amdgcn_s_wait_tensorcnt(0);
    }
    __syncthreads();                              // slab visible to all waves
#pragma unroll
    for (int u = 0; u < 6; ++u) {
      const int t = wv + 8 * u;                   // 0..47
      const int h = t >> 2, ct = t & 3;
      const half_t* pb = P + (size_t)h * L_ * L_ + (size_t)i0 * L_ + j0;
      acc[u] = wmma32(load_a_frag(pb, L_, lane),
                      load_b_frag(&us.sV[(h * DH_ + ct * 16) * 32], 32, lane),
                      acc[u]);
    }
    __syncthreads();                              // done with slab
  }
#pragma unroll
  for (int u = 0; u < 6; ++u) {
    const int t = wv + 8 * u;
    const int h = t >> 2, ct = t & 3;
#pragma unroll
    for (int r = 0; r < 8; ++r) us.yt[m0 + r][h * DH_ + ct * 16 + n] = acc[u][r];
  }
  __syncthreads();

  // residual + LayerNorm over D=768 per row; 16 threads/row, 48 elems each
  const int row = threadIdx.x >> 4, seg = threadIdx.x & 15;
  const int g  = s * L_ + i0 + row;
  const int d0 = seg * 48;
  float sum = 0.f, sq = 0.f;
  for (int u = 0; u < 48; ++u) {
    const float v = us.yt[row][d0 + u] + x[(size_t)g * D_ + d0 + u];
    us.yt[row][d0 + u] = v; sum += v; sq += v * v;
  }
  red[row][seg] = sum; __syncthreads();
  float S = 0.f; for (int k = 0; k < 16; ++k) S += red[row][k];
  __syncthreads();
  red[row][seg] = sq; __syncthreads();
  float Q = 0.f; for (int k = 0; k < 16; ++k) Q += red[row][k];
  const float mu = S / D_;
  const float rstd = rsqrtf(Q / D_ - mu * mu + EPS_);
  for (int u = 0; u < 48; ++u) {
    const int d = d0 + u;
    const float r = (us.yt[row][d] - mu) * rstd * g1[d] + b1[d];
    y[(size_t)g * D_ + d]  = r;
    yh[(size_t)g * D_ + d] = (half_t)r;
  }
}

// ---- Column attention, fully fused per (head h, position l).
// logits[i,j] = q[i,l,h,:].k[j,l,h,:]/8 + mask[j,l]; softmax over j; O = P@V.
__global__ __launch_bounds__(256) void k_col_attn(
    const half_t* __restrict__ qkv, const uint8_t* __restrict__ pm,
    float* __restrict__ colout) {
  __shared__ float  sm[64][64];
  __shared__ float  red[64][4];
  __shared__ half_t Pl[64][72];   // probs, f16, padded pitch (16B aligned)
  __shared__ half_t Vt[64][72];   // V transposed: Vt[c][j]
  const int lane = threadIdx.x & 31;
  const int wv   = threadIdx.x >> 5;
  const int l = blockIdx.x;
  const int h = blockIdx.y;
  const int n = lane & 15, m0 = (lane >> 4) * 8;
  const int ldq = L_ * TD_;       // row stride over the E index

  // logits: 16 tiles over 8 waves
#pragma unroll
  for (int u = 0; u < 2; ++u) {
    const int t = wv * 2 + u, it = t >> 2, jt = t & 3;
    v8f acc = {};
#pragma unroll
    for (int kc = 0; kc < DH_; kc += 32) {
      const half_t* qb = qkv + (size_t)(it * 16) * ldq + (size_t)l * TD_ + h * DH_ + kc;
      const half_t* kb = qkv + (size_t)(jt * 16) * ldq + (size_t)l * TD_ + D_ + h * DH_ + kc;
      acc = wmma32(load_a_frag(qb, ldq, lane), load_b_frag(kb, ldq, lane), acc);
    }
    const int j = jt * 16 + n;
    const float mk = pm[j * L_ + l] ? NEGV : 0.f;
#pragma unroll
    for (int r = 0; r < 8; ++r) sm[it * 16 + m0 + r][j] = acc[r] * 0.125f + mk;
  }
  // stage Vt[c][j] while logits settle (different LDS arrays, same barrier)
  {
    const int j = threadIdx.x >> 2, seg = threadIdx.x & 3;
    const half_t* vb = qkv + (size_t)j * ldq + (size_t)l * TD_ + 2 * D_ + h * DH_ + seg * 16;
    v8h v0 = *(const v8h*)vb;
    v8h v1 = *(const v8h*)(vb + 8);
#pragma unroll
    for (int c = 0; c < 8; ++c) {
      Vt[seg * 16 + c][j]     = v0[c];
      Vt[seg * 16 + 8 + c][j] = v1[c];
    }
  }
  __syncthreads();

  // softmax over j per row i; 4 threads/row, 16 cols each
  const int row = threadIdx.x >> 2, seg = threadIdx.x & 3;
  float lm = -3.0e38f;
  for (int c = 0; c < 16; ++c) lm = fmaxf(lm, sm[row][seg * 16 + c]);
  red[row][seg] = lm; __syncthreads();
  const float mx = fmaxf(fmaxf(red[row][0], red[row][1]),
                         fmaxf(red[row][2], red[row][3]));
  __syncthreads();
  float ls = 0.f;
  for (int c = 0; c < 16; ++c) {
    float e = __expf(sm[row][seg * 16 + c] - mx);
    sm[row][seg * 16 + c] = e; ls += e;
  }
  red[row][seg] = ls; __syncthreads();
  const float inv = 1.f / (red[row][0] + red[row][1] + red[row][2] + red[row][3]);
  for (int c = 0; c < 16; ++c)
    Pl[row][seg * 16 + c] = (half_t)(sm[row][seg * 16 + c] * inv);
  __syncthreads();

  // O = P @ V from LDS: 16 tiles over 8 waves, K=64
#pragma unroll
  for (int u = 0; u < 2; ++u) {
    const int t = wv * 2 + u, it = t >> 2, ct = t & 3;
    v8f acc = {};
#pragma unroll
    for (int j0 = 0; j0 < 64; j0 += 32)
      acc = wmma32(load_a_frag(&Pl[it * 16][0] + j0, 72, lane),
                   load_b_frag(&Vt[ct * 16][0] + j0, 72, lane), acc);
#pragma unroll
    for (int r = 0; r < 8; ++r) {
      const int i = it * 16 + m0 + r, c = ct * 16 + n;
      colout[(size_t)(i * L_ + l) * D_ + h * DH_ + c] = acc[r];
    }
  }
}

// ---- final residual + LayerNorm2 -> d_out -----------------------------------
__global__ __launch_bounds__(256) void k_ln2(
    const float* __restrict__ y, const float* __restrict__ co,
    const float* __restrict__ g2, const float* __restrict__ b2,
    float* __restrict__ out) {
  __shared__ float rs[256];
  const int g = blockIdx.x, t = threadIdx.x;
  float v[3]; float s = 0.f, q = 0.f;
#pragma unroll
  for (int e = 0; e < 3; ++e) {
    const int d = t + 256 * e;
    v[e] = y[(size_t)g * D_ + d] + co[(size_t)g * D_ + d];
    s += v[e]; q += v[e] * v[e];
  }
  rs[t] = s; __syncthreads();
  for (int o = 128; o > 0; o >>= 1) { if (t < o) rs[t] += rs[t + o]; __syncthreads(); }
  const float S = rs[0]; __syncthreads();
  rs[t] = q; __syncthreads();
  for (int o = 128; o > 0; o >>= 1) { if (t < o) rs[t] += rs[t + o]; __syncthreads(); }
  const float Q = rs[0];
  const float mu = S / D_;
  const float rstd = rsqrtf(Q / D_ - mu * mu + EPS_);
#pragma unroll
  for (int e = 0; e < 3; ++e) {
    const int d = t + 256 * e;
    out[(size_t)g * D_ + d] = (v[e] - mu) * rstd * g2[d] + b2[d];
  }
}

// ---------------------------------------------------------------------------

extern "C" void kernel_launch(void* const* d_in, const int* in_sizes, int n_in,
                              void* d_out, int out_size, void* d_ws, size_t ws_size,
                              hipStream_t stream) {
  (void)in_sizes; (void)n_in; (void)out_size; (void)ws_size;
  const float*   x     = (const float*)d_in[0];
  const uint8_t* pm    = (const uint8_t*)d_in[1];   // bool mask [E][L]
  const float*   w_row = (const float*)d_in[2];
  const float*   b_row = (const float*)d_in[3];
  const float*   w_col = (const float*)d_in[4];
  const float*   b_col = (const float*)d_in[5];
  const float*   g1    = (const float*)d_in[6];
  const float*   be1   = (const float*)d_in[7];
  const float*   g2    = (const float*)d_in[8];
  const float*   be2   = (const float*)d_in[9];

  float* out  = (float*)d_out;                 // [E][L][D] fp32
  float* maps = out + (size_t)N_ * D_;         // [H][1][L][L] fp32

  // workspace carve-up (~235 MB)
  char* ws = (char*)d_ws;
  size_t off = 0;
  auto take = [&](size_t bytes) -> char* {
    char* p = ws + off; off += (bytes + 255) & ~(size_t)255; return p;
  };
  half_t* xh   = (half_t*)take((size_t)N_ * D_ * 2);    // x f16; reused as yh
  half_t* wrh  = (half_t*)take((size_t)TD_ * D_ * 2);
  half_t* wch  = (half_t*)take((size_t)TD_ * D_ * 2);
  half_t* qkvh = (half_t*)take((size_t)N_ * TD_ * 2);   // reused for col pass
  half_t* vTh  = (half_t*)take((size_t)D_ * N_ * 2);
  half_t* Ph   = (half_t*)take((size_t)H_ * L_ * L_ * 2);
  float*  msum = (float*)take(L_ * 4);
  float*  y    = (float*)take((size_t)N_ * D_ * 4);
  float*  co   = (float*)take((size_t)N_ * D_ * 4);
  half_t* yh   = xh;  // xh is dead after the row QKV GEMM

  // precision drop for WMMA operands
  k_f32_to_f16<<<(N_ * D_ + 255) / 256, 256, 0, stream>>>(x, xh, N_ * D_);
  k_f32_to_f16<<<(TD_ * D_ + 255) / 256, 256, 0, stream>>>(w_row, wrh, TD_ * D_);
  k_f32_to_f16<<<(TD_ * D_ + 255) / 256, 256, 0, stream>>>(w_col, wch, TD_ * D_);
  k_masksum<<<1, 256, 0, stream>>>(pm, msum);

  // ---- row pass ----
  k_qkv_gemm<<<dim3(TD_ / 128, N_ / 128), 256, 0, stream>>>(xh, wrh, b_row,
                                                            qkvh, vTh, 1);
  k_row_attn_probs<<<dim3(L_ / 16, H_), 256, 0, stream>>>(qkvh, msum, maps, Ph);
  k_row_out_ln1<<<dim3(L_ / 16, E_), 256, 0, stream>>>(Ph, vTh, x, g1, be1, y, yh);

  // ---- column pass ----
  k_qkv_gemm<<<dim3(TD_ / 128, N_ / 128), 256, 0, stream>>>(yh, wch, b_col,
                                                            qkvh, vTh, 0);
  k_col_attn<<<dim3(L_, H_), 256, 0, stream>>>(qkvh, pm, co);
  k_ln2<<<N_, 256, 0, stream>>>(y, co, g2, be2, out);
}
